// ContrastiveLoss_2559800509023
// MI455X (gfx1250) — matrix operand order repaired and checked by reading, not compiled, not run on hardware
//
#include <hip/hip_runtime.h>
#include <hip/hip_bf16.h>
#include <math.h>

// NT-Xent contrastive loss for MI455X (gfx1250, wave32, WMMA).
// Pipeline: (1) normalize rows of z=[x;y] -> f16 zn in workspace,
//           (2) WMMA-tiled Gram matrix with fused exp/mask/row-sum,
//           (3) scalar loss reduction.

typedef __attribute__((ext_vector_type(16))) _Float16 v16h;
typedef __attribute__((ext_vector_type(8)))  _Float16 v8h;
typedef __attribute__((ext_vector_type(4)))  _Float16 v4h;
typedef __attribute__((ext_vector_type(8)))  float    v8f;

constexpr int N_ROWS = 4096;          // n
constexpr int D_DIM  = 1024;          // d
constexpr int M_ROWS = 8192;          // 2n
constexpr float INV_T_LOG2E = 9.61796693925975601f;  // log2(e)/0.15

// ---------------- Kernel 1: row normalize + f32->f16 ----------------
// One block per row; 256 threads x float4 covers d=1024.
__global__ void __launch_bounds__(256)
nt_xent_normalize(const float* __restrict__ x,
                  const float* __restrict__ y,
                  _Float16* __restrict__ zn) {
  const int row  = blockIdx.x;
  const int tid  = threadIdx.x;
  const int lane = tid & 31, wave = tid >> 5;
  const float* src = (row < N_ROWS) ? (x + (size_t)row * D_DIM)
                                    : (y + (size_t)(row - N_ROWS) * D_DIM);
  float4 v = ((const float4*)src)[tid];
  float ss = v.x*v.x + v.y*v.y + v.z*v.z + v.w*v.w;
  for (int off = 16; off > 0; off >>= 1) ss += __shfl_xor(ss, off, 32);
  __shared__ float red[8];
  __shared__ float s_scale;
  if (lane == 0) red[wave] = ss;
  __syncthreads();
  if (tid == 0) {
    float t = 0.f;
    #pragma unroll
    for (int w = 0; w < 8; ++w) t += red[w];
    s_scale = 1.f / fmaxf(sqrtf(t), 1e-8f);   // z / max(||z||, eps)
  }
  __syncthreads();
  const float scale = s_scale;
  v4h o;
  o[0] = (_Float16)(v.x * scale);
  o[1] = (_Float16)(v.y * scale);
  o[2] = (_Float16)(v.z * scale);
  o[3] = (_Float16)(v.w * scale);
  *(v4h*)(zn + (size_t)row * D_DIM + tid * 4) = o;
}

// ---------------- Kernel 2: WMMA Gram + fused exp/mask/rowsum ----------------
// Block = 8 waves; block tile = 128 rows x 1024 cols of aff.
// Wave owns 16 rows; 8 x v8f accumulators = 16x128 per K-sweep.
// aff never materialized: exp(s/T) computed in registers, diagonal and
// positive (j == i^n, valid since m=2n is a power of two) masked out.
__global__ void __launch_bounds__(256)
nt_xent_affinity(const _Float16* __restrict__ zn,
                 float* __restrict__ rowsum,
                 float* __restrict__ possum) {
  const int lane = threadIdx.x & 31;
  const int wave = threadIdx.x >> 5;
  const int h    = lane >> 4;        // which 16-lane half
  const int ln   = lane & 15;
  const int rowBase = blockIdx.x * 128 + wave * 16;
  const int colBlk  = blockIdx.y * 1024;

  // A fragment source: lane holds matrix row (lane%16).
  const _Float16* aBase = zn + (size_t)(rowBase + ln) * D_DIM;

  float rsum[8], pacc[8];
  #pragma unroll
  for (int r = 0; r < 8; ++r) { rsum[r] = 0.f; pacc[r] = 0.f; }

  for (int g = 0; g < 8; ++g) {
    const int colBase = colBlk + g * 128;
    v8f c[8];
    #pragma unroll
    for (int jj = 0; jj < 8; ++jj) c[jj] = (v8f){0,0,0,0,0,0,0,0};

    // B fragment source: B column (lane%16) of subtile jj == zn row.
    const _Float16* bBase[8];
    #pragma unroll
    for (int jj = 0; jj < 8; ++jj)
      bBase[jj] = zn + (size_t)(colBase + jj * 16 + ln) * D_DIM + 16 * h;

    for (int kb = 0; kb < D_DIM; kb += 32) {
      // A 16x32 f16 layout: lane holds K in [16h, 16h+16) contiguous.
      union { v16h v; v8h c8[2]; } a;
      a.c8[0] = *(const v8h*)(aBase + kb + 16 * h);
      a.c8[1] = *(const v8h*)(aBase + kb + 16 * h + 8);
      #pragma unroll
      for (int jj = 0; jj < 8; ++jj) {
        // B 32x16 f16 layout: lane holds K in [16h, 16h+16) contiguous.
        v16h b = *(const v16h*)(bBase[jj] + kb);
        c[jj] = __builtin_amdgcn_wmma_f32_16x16x32_f16(
            false, a.v, false, b, (short)0, c[jj], false, false);
      }
    }

    // Fused epilogue: e = exp(s/T); mask diag and positive pair.
    #pragma unroll
    for (int jj = 0; jj < 8; ++jj) {
      const int gj = colBase + jj * 16 + ln;
      #pragma unroll
      for (int r = 0; r < 8; ++r) {
        const int gi = rowBase + 8 * h + r;     // C layout: row = r + 8h
        float e = exp2f(c[jj][r] * INV_T_LOG2E);
        bool ispos = (gj == (gi ^ N_ROWS));     // (i+n) % 2n == i^n
        bool skip  = ispos || (gj == gi);
        rsum[r] += skip ? 0.f : e;
        pacc[r] += ispos ? e : 0.f;
      }
    }
  }

  // Reduce across the 16 lanes holding each row, then one atomic per row.
  #pragma unroll
  for (int r = 0; r < 8; ++r) {
    float rv = rsum[r], pv = pacc[r];
    for (int off = 1; off < 16; off <<= 1) {
      rv += __shfl_xor(rv, off, 16);
      pv += __shfl_xor(pv, off, 16);
    }
    if (ln == 0) {
      const int gi = rowBase + 8 * h + r;
      atomicAdd(&rowsum[gi], rv);
      atomicAdd(&possum[gi], pv);
    }
  }
}

// ---------------- Kernel 3: loss = sum(log(neg) - log(pos)) ----------------
__global__ void __launch_bounds__(256)
nt_xent_loss(const float* __restrict__ rowsum,
             const float* __restrict__ possum,
             float* __restrict__ out) {
  float local = 0.f;
  for (int i = threadIdx.x; i < M_ROWS; i += 256)
    local += logf(rowsum[i]) - logf(possum[i]);
  for (int off = 16; off > 0; off >>= 1) local += __shfl_xor(local, off, 32);
  __shared__ float red[8];
  const int lane = threadIdx.x & 31, wave = threadIdx.x >> 5;
  if (lane == 0) red[wave] = local;
  __syncthreads();
  if (threadIdx.x == 0) {
    float t = 0.f;
    #pragma unroll
    for (int w = 0; w < 8; ++w) t += red[w];
    out[0] = t;
  }
}

extern "C" void kernel_launch(void* const* d_in, const int* in_sizes, int n_in,
                              void* d_out, int out_size, void* d_ws, size_t ws_size,
                              hipStream_t stream) {
  const float* x = (const float*)d_in[0];
  const float* y = (const float*)d_in[1];

  // Workspace layout: zn f16 [8192*1024] (16 MB), then rowsum/possum f32 [8192] each.
  _Float16* zn = (_Float16*)d_ws;
  float* rowsum = (float*)((char*)d_ws + (size_t)M_ROWS * D_DIM * sizeof(_Float16));
  float* possum = rowsum + M_ROWS;

  hipMemsetAsync(rowsum, 0, 2 * (size_t)M_ROWS * sizeof(float), stream);
  nt_xent_normalize<<<M_ROWS, 256, 0, stream>>>(x, y, zn);
  nt_xent_affinity<<<dim3(M_ROWS / 128, 8), 256, 0, stream>>>(zn, rowsum, possum);
  nt_xent_loss<<<1, 256, 0, stream>>>(rowsum, possum, (float*)d_out);
}
